// ROIGather_87136296501619
// MI455X (gfx1250) — matrix operand order repaired and checked by reading, not compiled, not run on hardware
//
#include <hip/hip_runtime.h>
#include <hip/hip_bf16.h>
#include <math.h>

// ---------------------------------------------------------------------------
// Types / WMMA helpers (CDNA5 gfx1250, wave32)
// ---------------------------------------------------------------------------
typedef __attribute__((ext_vector_type(16))) __bf16 v16bf;
typedef __attribute__((ext_vector_type(8)))  float  v8f;

union BF16Frag { v16bf v; unsigned u[8]; };

#define BNS 0.99999500003749937f   /* 1/sqrt(1+1e-5) */

__device__ __forceinline__ v8f vzero8() {
  v8f z;
#pragma unroll
  for (int i = 0; i < 8; ++i) z[i] = 0.f;
  return z;
}

__device__ __forceinline__ v8f wmma_bf16(v16bf a, v16bf b, v8f c) {
  return __builtin_amdgcn_wmma_f32_16x16x32_bf16(false, a, false, b, (short)0, c,
                                                 false, false);
}

// A-matrix fragment: 16x32 bf16, element (m,k) at p[m*ld + k].
// Lane L: m=L&15, half=L>>4. VGPR j holds K = (j>=4?16:0) + half*8 + 2*(j&3) {pair}.
__device__ __forceinline__ v16bf load_frag_a(const __bf16* p, int lane, int ld) {
  const int m = lane & 15, half = lane >> 4;
  const __bf16* row = p + m * ld + half * 8;
  BF16Frag f;
#pragma unroll
  for (int j = 0; j < 8; ++j) {
    int k = ((j >> 2) << 4) + ((j & 3) << 1);
    f.u[j] = *reinterpret_cast<const unsigned*>(row + k);
  }
  return f.v;
}

// B-matrix fragment: 32x16 bf16, element (k,n) at p[n*ld + k] ("[n][k]" layout).
// Lane L: n=L&15, half=L>>4. VGPR j holds K = (j>=4?16:0) + half*8 + 2*(j&3) {pair}.
__device__ __forceinline__ v16bf load_frag_b(const __bf16* p, int lane, int ld) {
  const int n = lane & 15, half = lane >> 4;
  const __bf16* row = p + n * ld + half * 8;
  BF16Frag f;
#pragma unroll
  for (int j = 0; j < 8; ++j)
    f.u[j] = *reinterpret_cast<const unsigned*>(row + ((j >> 2) << 4) + ((j & 3) << 1));
  return f.v;
}

__device__ __forceinline__ float sigm(float x) { return 1.f / (1.f + expf(-x)); }

// Problem constants
#define NB   16
#define NP_  192
#define SPAT 3060      /* 68*45 */
#define CC   64
#define MIDC 48

// ---------------------------------------------------------------------------
// Weight prep: fp32 -> bf16 relayouts
// ---------------------------------------------------------------------------
__global__ void k_f32_to_bf16(const float* __restrict__ s, __bf16* __restrict__ d, int n) {
  int i = blockIdx.x * 256 + threadIdx.x;
  if (i < n) d[i] = (__bf16)s[i];
}

// wc[l][o][dk][c] <- conv_wl[o][c][dk]   (3,48,9,64)
__global__ void k_prep_convw(const float* w0, const float* w1, const float* w2,
                             __bf16* wc) {
  int i = blockIdx.x * 256 + threadIdx.x;
  if (i >= 3 * 48 * 9 * 64) return;
  int c = i & 63;
  int t = i >> 6;
  int dk = t % 9; t /= 9;
  int o = t % 48;
  int l = t / 48;
  const float* w = (l == 0) ? w0 : (l == 1) ? w1 : w2;
  wc[i] = (__bf16)w[(o * 64 + c) * 9 + dk];
}

// wcat[o][dk][c160] <- cat_w[o][c][dk] (zero pad c>=144)
__global__ void k_prep_cat(const float* catw, __bf16* wcat) {
  int i = blockIdx.x * 256 + threadIdx.x;
  if (i >= 64 * 9 * 160) return;
  int c = i % 160;
  int t = i / 160;
  int dk = t % 9;
  int o = t / 9;
  wcat[i] = (c < 144) ? (__bf16)catw[(o * 144 + c) * 9 + dk] : (__bf16)0.f;
}

// ---------------------------------------------------------------------------
// ODConv coefficients: A[b][o][c] = fl[o]*ch[c]*sum_k kn_k W[k][o][c]  (bf16)
// ---------------------------------------------------------------------------
__global__ __launch_bounds__(256) void k_odcoef(
    const float* __restrict__ X, const float* fcw, const float* bng, const float* bnb,
    const float* chw, const float* chb, const float* flw, const float* flb,
    const float* knw, const float* knb, const float* W, __bf16* __restrict__ Aod) {
  __shared__ float red[256];
  __shared__ float pooled[64], a16[16], chs[64], fls[64], kns[4];
  int b = blockIdx.x, tid = threadIdx.x;
  const float* Xb = X + (size_t)b * (CC * SPAT);
  int c = tid >> 2, sub = tid & 3;
  float s = 0.f;
  for (int p = sub; p < SPAT; p += 4) s += Xb[c * SPAT + p];
  red[tid] = s;
  __syncthreads();
  if (sub == 0)
    pooled[c] = (red[tid] + red[tid + 1] + red[tid + 2] + red[tid + 3]) * (1.f / SPAT);
  __syncthreads();
  if (tid < 16) {
    float t = 0.f;
    for (int k = 0; k < 64; ++k) t += pooled[k] * fcw[tid * 64 + k];
    a16[tid] = fmaxf(t * bng[tid] * BNS + bnb[tid], 0.f);
  }
  __syncthreads();
  if (tid < 64) {
    float t1 = 0.f, t2 = 0.f;
    for (int k = 0; k < 16; ++k) {
      t1 += a16[k] * chw[tid * 16 + k];
      t2 += a16[k] * flw[tid * 16 + k];
    }
    chs[tid] = sigm(t1 + chb[tid]);
    fls[tid] = sigm(t2 + flb[tid]);
  }
  if (tid >= 64 && tid < 68) {
    int q = tid - 64;
    float t = 0.f;
    for (int k = 0; k < 16; ++k) t += a16[k] * knw[q * 16 + k];
    kns[q] = t + knb[q];
  }
  __syncthreads();
  if (tid == 0) {
    float mx = fmaxf(fmaxf(kns[0], kns[1]), fmaxf(kns[2], kns[3]));
    float e0 = expf(kns[0] - mx), e1 = expf(kns[1] - mx),
          e2 = expf(kns[2] - mx), e3 = expf(kns[3] - mx);
    float inv = 1.f / (e0 + e1 + e2 + e3);
    kns[0] = e0 * inv; kns[1] = e1 * inv; kns[2] = e2 * inv; kns[3] = e3 * inv;
  }
  __syncthreads();
  for (int idx = tid; idx < 4096; idx += 256) {
    int o = idx >> 6, ci = idx & 63;
    float w = kns[0] * W[idx] + kns[1] * W[4096 + idx] + kns[2] * W[8192 + idx] +
              kns[3] * W[12288 + idx];
    Aod[(size_t)b * 4096 + idx] = (__bf16)(fls[o] * chs[ci] * w);
  }
}

// ---------------------------------------------------------------------------
// Channel-mix GEMM: Y[b] = epi( A_b(64x64) @ X[b](64xN) ), WMMA bf16
// mode 0: OD   (N=3060, out fp32 [b][o][pos])
// mode 1: KEY  (gathered 250, out bf16 [b][pos][o], epi relu(bn))
// mode 2: VAL  (gathered 250, out bf16 [b][o][pos256], epi +bias)
// ---------------------------------------------------------------------------
__global__ __launch_bounds__(128) void k_chanmix(
    const float* __restrict__ X, const __bf16* __restrict__ Amat, int Aper,
    float* __restrict__ outF, __bf16* __restrict__ outB,
    const float* eg, const float* eb, int mode) {
  __shared__ __align__(16) __bf16 xt[16 * 64];  // [i16][c64]
  int nt = blockIdx.x, b = blockIdx.y, tid = threadIdx.x;
  for (int s2 = tid; s2 < 1024; s2 += 128) {
    int c = s2 >> 4, i = s2 & 15;
    int j = nt * 16 + i;
    float v = 0.f;
    if (mode == 0) {
      if (j < SPAT) v = X[(size_t)b * (CC * SPAT) + c * SPAT + j];
    } else {
      if (j < 250) {
        int r10 = j / 25, c25 = j - r10 * 25;
        int sp = ((r10 * 68) / 10) * 45 + (c25 * 45) / 25;
        v = X[(size_t)b * (CC * SPAT) + c * SPAT + sp];
      }
    }
    xt[i * 64 + c] = (__bf16)v;
  }
  __syncthreads();
  int wave = tid >> 5, lane = tid & 31;
  const __bf16* A = Amat + (size_t)b * Aper;
  v8f acc = vzero8();
#pragma unroll
  for (int kc = 0; kc < 2; ++kc) {
    v16bf af = load_frag_a(A + wave * 16 * 64 + kc * 32, lane, 64);
    v16bf bf = load_frag_b(xt + kc * 32, lane, 64);
    acc = wmma_bf16(af, bf, acc);
  }
  int nloc = lane & 15, half = lane >> 4;
  int pos = nt * 16 + nloc;
  if (mode == 0) {
    if (pos < SPAT) {
#pragma unroll
      for (int r = 0; r < 8; ++r) {
        int o = wave * 16 + half * 8 + r;
        outF[(size_t)b * (CC * SPAT) + (size_t)o * SPAT + pos] = acc[r];
      }
    }
  } else if (mode == 1) {
#pragma unroll
    for (int r = 0; r < 8; ++r) {
      int o = wave * 16 + half * 8 + r;
      float v = fmaxf(acc[r] * eg[o] * BNS + eb[o], 0.f);
      outB[(size_t)b * 16384 + pos * 64 + o] = (__bf16)v;
    }
  } else {
#pragma unroll
    for (int r = 0; r < 8; ++r) {
      int o = wave * 16 + half * 8 + r;
      outB[(size_t)b * 16384 + o * 256 + pos] = (__bf16)(acc[r] + eb[o]);
    }
  }
}

// ---------------------------------------------------------------------------
// scSE: pooled MLP (per batch), then elementwise apply
// ---------------------------------------------------------------------------
__global__ __launch_bounds__(256) void k_scsepool(const float* __restrict__ X,
                                                  const float* sqw, const float* exw,
                                                  float* __restrict__ zout) {
  __shared__ float red[256];
  __shared__ float pooled[64], h[32];
  int b = blockIdx.x, tid = threadIdx.x;
  const float* Xb = X + (size_t)b * (CC * SPAT);
  int c = tid >> 2, sub = tid & 3;
  float s = 0.f;
  for (int p = sub; p < SPAT; p += 4) s += Xb[c * SPAT + p];
  red[tid] = s;
  __syncthreads();
  if (sub == 0)
    pooled[c] = (red[tid] + red[tid + 1] + red[tid + 2] + red[tid + 3]) * (1.f / SPAT);
  __syncthreads();
  if (tid < 32) {
    float t = 0.f;
    for (int k = 0; k < 64; ++k) t += pooled[k] * sqw[tid * 64 + k];
    h[tid] = fmaxf(t, 0.f);
  }
  __syncthreads();
  if (tid < 64) {
    float t = 0.f;
    for (int k = 0; k < 32; ++k) t += h[k] * exw[tid * 32 + k];
    zout[b * 64 + tid] = sigm(t);
  }
}

__global__ __launch_bounds__(256) void k_scseapply(const float* __restrict__ X,
                                                   const float* __restrict__ z,
                                                   const float* sw,
                                                   float* __restrict__ Y) {
  int gid = blockIdx.x * 256 + threadIdx.x;
  if (gid >= NB * SPAT) return;
  int b = gid / SPAT, pos = gid - b * SPAT;
  const float* Xb = X + (size_t)b * (CC * SPAT) + pos;
  float q = 0.f;
  for (int c = 0; c < 64; ++c) q += Xb[(size_t)c * SPAT] * sw[c];
  q = sigm(q);
  float* Yb = Y + (size_t)b * (CC * SPAT) + pos;
  for (int c = 0; c < 64; ++c)
    Yb[(size_t)c * SPAT] = Xb[(size_t)c * SPAT] * (z[b * 64 + c] + q);
}

// ---------------------------------------------------------------------------
// Fused ROI conv pipeline per sample: 3x conv(9,1)+BN+ReLU -> concat (LDS)
// -> cat conv(9,1)+BN+ReLU -> roiflat bf16 [n][c*36+s]. All GEMMs via WMMA.
// ---------------------------------------------------------------------------
__global__ __launch_bounds__(288) void k_roiconv(
    const float* __restrict__ roi0, const float* __restrict__ roi1,
    const float* __restrict__ roi2, const __bf16* __restrict__ wc,
    const __bf16* __restrict__ wcat, const float* g0, const float* b0,
    const float* g1, const float* b1, const float* g2, const float* b2,
    const float* cg, const float* cb, __bf16* __restrict__ roiflat) {
  __shared__ __align__(16) __bf16 xin[3 * 56 * 64];   // [l][sp56][c64], halo zero
  __shared__ __align__(16) __bf16 feats[56 * 160];    // [sp56][c160], halo+pad zero
  int n = blockIdx.x, tid = threadIdx.x;
  // Stage inputs (with 4-row zero halo on each side)
  for (int s = tid; s < 3 * 56 * 64; s += 288) {
    int l = s / (56 * 64);
    int rem = s - l * (56 * 64);
    int sp = rem >> 6;
    int c = rem & 63;
    float v = 0.f;
    int srow = sp - 4;
    if (srow >= 0 && srow < 36) {
      const float* src = (l == 0) ? roi0 : (l == 1) ? roi1 : roi2;
      v = src[(size_t)n * 2304 + c * 36 + srow];
    }
    xin[s] = (__bf16)v;
  }
  for (int s = tid; s < 56 * 160; s += 288) feats[s] = (__bf16)0.f;
  __syncthreads();

  int wave = tid >> 5, lane = tid & 31;
  int half = lane >> 4;
  // Stage 2: per-layer convs -> feats LDS. wave = layer*3 + mtile
  {
    int l = wave / 3, mloc = wave % 3;
    const float* gs = (l == 0) ? g0 : (l == 1) ? g1 : g2;
    const float* bs = (l == 0) ? b0 : (l == 1) ? b1 : b2;
    const __bf16* wl = wc + l * (48 * 9 * 64);
    for (int nt = 0; nt < 3; ++nt) {
      v8f acc = vzero8();
      for (int dk = 0; dk < 9; ++dk) {
#pragma unroll
        for (int kc = 0; kc < 2; ++kc) {
          v16bf af = load_frag_a(wl + (mloc * 144 + dk) * 64 + kc * 32, lane, 576);
          v16bf bf = load_frag_b(xin + l * 3584 + (nt * 16 + dk) * 64 + kc * 32,
                                 lane, 64);
          acc = wmma_bf16(af, bf, acc);
        }
      }
      int sout = nt * 16 + (lane & 15);
      if (sout < 36) {
        int sp = sout + 4;
#pragma unroll
        for (int r = 0; r < 8; ++r) {
          int o = mloc * 16 + half * 8 + r;
          float v = fmaxf(acc[r] * gs[o] * BNS + bs[o], 0.f);
          feats[sp * 160 + l * 48 + o] = (__bf16)v;
        }
      }
    }
  }
  __syncthreads();
  // Stage 3: cat conv (64 out x 36), 12 tiles over 9 waves
  for (int t = wave; t < 12; t += 9) {
    int mt = t & 3, ntile = t >> 2;
    v8f acc = vzero8();
    for (int dk = 0; dk < 9; ++dk) {
#pragma unroll
      for (int kc = 0; kc < 5; ++kc) {
        v16bf af = load_frag_a(wcat + (mt * 144 + dk) * 160 + kc * 32, lane, 1440);
        v16bf bf = load_frag_b(feats + (ntile * 16 + dk) * 160 + kc * 32, lane, 160);
        acc = wmma_bf16(af, bf, acc);
      }
    }
    int sout = ntile * 16 + (lane & 15);
    if (sout < 36) {
#pragma unroll
      for (int r = 0; r < 8; ++r) {
        int o = mt * 16 + half * 8 + r;
        float v = fmaxf(acc[r] * cg[o] * BNS + cb[o], 0.f);
        roiflat[(size_t)n * 2304 + o * 36 + sout] = (__bf16)v;
      }
    }
  }
}

// ---------------------------------------------------------------------------
// fc GEMM: (3072 x 2304) @ (2304 x 64), relu(+bias) -> fcout fp32
// ---------------------------------------------------------------------------
__global__ __launch_bounds__(128) void k_fc(const __bf16* __restrict__ roiflat,
                                            const __bf16* __restrict__ wfc,
                                            const float* fcb,
                                            float* __restrict__ fcout) {
  int mt = blockIdx.x;
  int wave = threadIdx.x >> 5, lane = threadIdx.x & 31;
  v8f acc = vzero8();
  for (int kc = 0; kc < 72; ++kc) {
    v16bf af = load_frag_a(roiflat + (size_t)mt * 16 * 2304 + kc * 32, lane, 2304);
    v16bf bf = load_frag_b(wfc + wave * 16 * 2304 + kc * 32, lane, 2304);
    acc = wmma_bf16(af, bf, acc);
  }
  int col = wave * 16 + (lane & 15), half = lane >> 4;
  float bias = fcb[col];
#pragma unroll
  for (int r = 0; r < 8; ++r) {
    int row = mt * 16 + half * 8 + r;
    fcout[(size_t)row * 64 + col] = fmaxf(acc[r] + bias, 0.f);
  }
}

// ---------------------------------------------------------------------------
// LayerNorm over C=64 + query = relu(roi*q_w[p]+q_b[p]) (bf16)
// ---------------------------------------------------------------------------
__global__ __launch_bounds__(64) void k_ln(const float* __restrict__ fcout,
                                           const float* lg, const float* lb,
                                           const float* qw, const float* qb,
                                           float* __restrict__ roiLN,
                                           __bf16* __restrict__ query) {
  __shared__ float red[64];
  int row = blockIdx.x, tid = threadIdx.x;
  float v = fcout[(size_t)row * 64 + tid];
  red[tid] = v;
  __syncthreads();
  for (int s = 32; s > 0; s >>= 1) {
    if (tid < s) red[tid] += red[tid + s];
    __syncthreads();
  }
  float mu = red[0] * (1.f / 64.f);
  __syncthreads();
  float d = v - mu;
  red[tid] = d * d;
  __syncthreads();
  for (int s = 32; s > 0; s >>= 1) {
    if (tid < s) red[tid] += red[tid + s];
    __syncthreads();
  }
  float var = red[0] * (1.f / 64.f);
  float y = d * rsqrtf(var + 1e-5f) * lg[tid] + lb[tid];
  roiLN[(size_t)row * 64 + tid] = y;
  int p = row % NP_;
  query[(size_t)row * 64 + tid] = (__bf16)fmaxf(y * qw[p] + qb[p], 0.f);
}

// ---------------------------------------------------------------------------
// Attention scores: (192 x 64) @ (64 x 256) per batch, *C^-0.5 -> fp32
// ---------------------------------------------------------------------------
__global__ __launch_bounds__(512) void k_scores(const __bf16* __restrict__ query,
                                                const __bf16* __restrict__ keyf,
                                                float* __restrict__ scores) {
  int mt = blockIdx.x, b = blockIdx.y;
  int wave = threadIdx.x >> 5, lane = threadIdx.x & 31;
  v8f acc = vzero8();
#pragma unroll
  for (int kc = 0; kc < 2; ++kc) {
    v16bf af = load_frag_a(query + ((size_t)b * NP_ + mt * 16) * 64 + kc * 32, lane, 64);
    v16bf bf = load_frag_b(keyf + ((size_t)b * 256 + wave * 16) * 64 + kc * 32, lane, 64);
    acc = wmma_bf16(af, bf, acc);
  }
  int nn = wave * 16 + (lane & 15), half = lane >> 4;
#pragma unroll
  for (int r = 0; r < 8; ++r) {
    int p = mt * 16 + half * 8 + r;
    scores[((size_t)b * NP_ + p) * 256 + nn] = acc[r] * 0.125f;
  }
}

// softmax over n<250, pad cols -> 0 (bf16 sim)
__global__ __launch_bounds__(256) void k_softmax(const float* __restrict__ scores,
                                                 __bf16* __restrict__ sim) {
  __shared__ float red[256];
  int row = blockIdx.x, tid = threadIdx.x;
  const float* sr = scores + (size_t)row * 256;
  float v = (tid < 250) ? sr[tid] : -1e30f;
  red[tid] = v;
  __syncthreads();
  for (int s = 128; s > 0; s >>= 1) {
    if (tid < s) red[tid] = fmaxf(red[tid], red[tid + s]);
    __syncthreads();
  }
  float mx = red[0];
  __syncthreads();
  float e = (tid < 250) ? expf(v - mx) : 0.f;
  red[tid] = e;
  __syncthreads();
  for (int s = 128; s > 0; s >>= 1) {
    if (tid < s) red[tid] += red[tid + s];
    __syncthreads();
  }
  sim[(size_t)row * 256 + tid] = (__bf16)(e / red[0]);
}

// ctx GEMM: (192 x 256) @ (256 x 64) per batch + residual -> d_out fp32
__global__ __launch_bounds__(128) void k_ctx(const __bf16* __restrict__ sim,
                                             const __bf16* __restrict__ value,
                                             const float* __restrict__ roiLN,
                                             const float* Ww, const float* Wb,
                                             float* __restrict__ out) {
  int mt = blockIdx.x, b = blockIdx.y;
  int wave = threadIdx.x >> 5, lane = threadIdx.x & 31;
  v8f acc = vzero8();
#pragma unroll
  for (int kc = 0; kc < 8; ++kc) {
    v16bf af = load_frag_a(sim + ((size_t)b * NP_ + mt * 16) * 256 + kc * 32, lane, 256);
    v16bf bf = load_frag_b(value + ((size_t)b * 64 + wave * 16) * 256 + kc * 32, lane, 256);
    acc = wmma_bf16(af, bf, acc);
  }
  int c = wave * 16 + (lane & 15), half = lane >> 4;
#pragma unroll
  for (int r = 0; r < 8; ++r) {
    int p = mt * 16 + half * 8 + r;
    size_t i = ((size_t)b * NP_ + p) * 64 + c;
    out[i] = roiLN[i] + acc[r] * Ww[p] + Wb[p];
  }
}

// ---------------------------------------------------------------------------
// Host launcher
// ---------------------------------------------------------------------------
extern "C" void kernel_launch(void* const* d_in, const int* in_sizes, int n_in,
                              void* d_out, int out_size, void* d_ws, size_t ws_size,
                              hipStream_t stream) {
  const float* roi0 = (const float*)d_in[0];
  const float* roi1 = (const float*)d_in[1];
  const float* roi2 = (const float*)d_in[2];
  const float* x    = (const float*)d_in[3];
  const float* od_fc_w = (const float*)d_in[4];
  const float* od_bn_g = (const float*)d_in[5];
  const float* od_bn_b = (const float*)d_in[6];
  const float* od_ch_w = (const float*)d_in[7];
  const float* od_ch_b = (const float*)d_in[8];
  const float* od_fl_w = (const float*)d_in[9];
  const float* od_fl_b = (const float*)d_in[10];
  const float* od_kn_w = (const float*)d_in[11];
  const float* od_kn_b = (const float*)d_in[12];
  const float* od_weight = (const float*)d_in[13];
  const float* cse_sq_w = (const float*)d_in[14];
  const float* cse_ex_w = (const float*)d_in[15];
  const float* sse_w = (const float*)d_in[16];
  const float* conv_w0 = (const float*)d_in[17];
  const float* conv_g0 = (const float*)d_in[18];
  const float* conv_b0 = (const float*)d_in[19];
  const float* conv_w1 = (const float*)d_in[20];
  const float* conv_g1 = (const float*)d_in[21];
  const float* conv_b1 = (const float*)d_in[22];
  const float* conv_w2 = (const float*)d_in[23];
  const float* conv_g2 = (const float*)d_in[24];
  const float* conv_b2 = (const float*)d_in[25];
  const float* cat_w = (const float*)d_in[26];
  const float* cat_g = (const float*)d_in[27];
  const float* cat_b = (const float*)d_in[28];
  const float* fc_w = (const float*)d_in[29];
  const float* fc_b = (const float*)d_in[30];
  const float* ln_g = (const float*)d_in[31];
  const float* ln_b = (const float*)d_in[32];
  const float* q_w = (const float*)d_in[33];
  const float* q_b = (const float*)d_in[34];
  const float* key_w = (const float*)d_in[35];
  const float* key_g = (const float*)d_in[36];
  const float* key_b = (const float*)d_in[37];
  const float* val_w = (const float*)d_in[38];
  const float* val_b = (const float*)d_in[39];
  const float* Ww = (const float*)d_in[40];
  const float* Wb = (const float*)d_in[41];
  (void)in_sizes; (void)n_in; (void)out_size; (void)ws_size;

  char* wp = (char*)d_ws;
  auto carve = [&](size_t bytes) -> void* {
    void* p = (void*)wp;
    wp += (bytes + 255) & ~(size_t)255;
    return p;
  };
  float*  xA      = (float*)carve((size_t)NB * CC * SPAT * 4);
  float*  xB      = (float*)carve((size_t)NB * CC * SPAT * 4);
  __bf16* Aod     = (__bf16*)carve((size_t)NB * 4096 * 2);
  float*  zbuf    = (float*)carve((size_t)NB * 64 * 4);
  __bf16* wc      = (__bf16*)carve((size_t)3 * 48 * 9 * 64 * 2);
  __bf16* wcat    = (__bf16*)carve((size_t)64 * 9 * 160 * 2);
  __bf16* wfc     = (__bf16*)carve((size_t)64 * 2304 * 2);
  __bf16* kwb     = (__bf16*)carve((size_t)4096 * 2);
  __bf16* vwb     = (__bf16*)carve((size_t)4096 * 2);
  __bf16* roiflat = (__bf16*)carve((size_t)3072 * 2304 * 2);
  float*  fcout   = (float*)carve((size_t)3072 * 64 * 4);
  float*  roiLN   = (float*)carve((size_t)3072 * 64 * 4);
  __bf16* query   = (__bf16*)carve((size_t)3072 * 64 * 2);
  __bf16* keyf    = (__bf16*)carve((size_t)NB * 256 * 64 * 2);
  __bf16* value   = (__bf16*)carve((size_t)NB * 64 * 256 * 2);
  float*  scores  = (float*)carve((size_t)NB * NP_ * 256 * 4);
  __bf16* sim     = (__bf16*)carve((size_t)NB * NP_ * 256 * 2);

  // --- weight prep ---
  k_prep_convw<<<(3 * 48 * 9 * 64 + 255) / 256, 256, 0, stream>>>(conv_w0, conv_w1,
                                                                  conv_w2, wc);
  k_prep_cat<<<(64 * 9 * 160 + 255) / 256, 256, 0, stream>>>(cat_w, wcat);
  k_f32_to_bf16<<<(64 * 2304 + 255) / 256, 256, 0, stream>>>(fc_w, wfc, 64 * 2304);
  k_f32_to_bf16<<<16, 256, 0, stream>>>(key_w, kwb, 4096);
  k_f32_to_bf16<<<16, 256, 0, stream>>>(val_w, vwb, 4096);

  // --- od #1 -> xA ---
  k_odcoef<<<NB, 256, 0, stream>>>(x, od_fc_w, od_bn_g, od_bn_b, od_ch_w, od_ch_b,
                                   od_fl_w, od_fl_b, od_kn_w, od_kn_b, od_weight, Aod);
  k_chanmix<<<dim3(192, NB), 128, 0, stream>>>(x, Aod, 4096, xA, nullptr, nullptr,
                                               nullptr, 0);
  // --- scSE -> xB ---
  k_scsepool<<<NB, 256, 0, stream>>>(xA, cse_sq_w, cse_ex_w, zbuf);
  k_scseapply<<<(NB * SPAT + 255) / 256, 256, 0, stream>>>(xA, zbuf, sse_w, xB);
  // --- od #2 -> xA ---
  k_odcoef<<<NB, 256, 0, stream>>>(xB, od_fc_w, od_bn_g, od_bn_b, od_ch_w, od_ch_b,
                                   od_fl_w, od_fl_b, od_kn_w, od_kn_b, od_weight, Aod);
  k_chanmix<<<dim3(192, NB), 128, 0, stream>>>(xB, Aod, 4096, xA, nullptr, nullptr,
                                               nullptr, 0);
  // --- ROI conv pipeline ---
  k_roiconv<<<3072, 288, 0, stream>>>(roi0, roi1, roi2, wc, wcat, conv_g0, conv_b0,
                                      conv_g1, conv_b1, conv_g2, conv_b2, cat_g,
                                      cat_b, roiflat);
  k_fc<<<192, 128, 0, stream>>>(roiflat, wfc, fc_b, fcout);
  k_ln<<<3072, 64, 0, stream>>>(fcout, ln_g, ln_b, q_w, q_b, roiLN, query);
  // --- key/value projections on gathered 250 positions ---
  k_chanmix<<<dim3(16, NB), 128, 0, stream>>>(xA, kwb, 0, nullptr, keyf, key_g,
                                              key_b, 1);
  k_chanmix<<<dim3(16, NB), 128, 0, stream>>>(xA, vwb, 0, nullptr, value, nullptr,
                                              val_b, 2);
  // --- attention ---
  k_scores<<<dim3(12, NB), 512, 0, stream>>>(query, keyf, scores);
  k_softmax<<<NB * NP_, 256, 0, stream>>>(scores, sim);
  k_ctx<<<dim3(12, NB), 128, 0, stream>>>(sim, value, roiLN, Ww, Wb, (float*)d_out);
}
